// DAGG_NAPLGraph_44805098832342
// MI455X (gfx1250) — matrix operand order repaired and checked by reading, not compiled, not run on hardware
//
#include <hip/hip_runtime.h>
#include <hip/hip_bf16.h>

typedef __attribute__((ext_vector_type(16))) _Float16 v16h;
typedef __attribute__((ext_vector_type(8)))  _Float16 v8h;
typedef __attribute__((ext_vector_type(8)))  float    v8f;

#define B_   32
#define L_   1024
#define DIN_ 256
#define EMB_ 128
#define OUT_ 128

// Load a WMMA A-fragment (16x32 f16): per-lane one contiguous 16-half run at
// K = 2*kb..2*kb+15, kb = 0 (lanes 0-15) or 8 (lanes 16-31).
__device__ __forceinline__ v16h ldsA(const _Float16* rowbase, int kb) {
  return *(const v16h*)(rowbase + 2 * kb);
}

// Load a WMMA B-fragment (32x16 f16): per-lane one contiguous 16-half run,
// N = lane%16, K base = 0 (lanes 0-15) or 16 (lanes 16-31).
__device__ __forceinline__ v16h ldsB(const _Float16* p) {
  return *(const v16h*)p;
}

// ---------------------------------------------------------------------------
// Kernel 0: fold W_eg @ Wg into one weight, transpose weights to f16 [n][k]
// layout so WMMA B-fragments are contiguous-K loads per lane.
// ---------------------------------------------------------------------------
__global__ __launch_bounds__(128) void prep_weights(
    const float* __restrict__ W_ge, const float* __restrict__ W_eg,
    const float* __restrict__ b_eg, const float* __restrict__ Wg,
    _Float16* __restrict__ WgeT, _Float16* __restrict__ WcT,
    float* __restrict__ bc) {
  const int k = blockIdx.x;   // 0..DIN-1
  const int d = threadIdx.x;  // 0..127
  WgeT[d * DIN_ + k] = (_Float16)W_ge[k * EMB_ + d];
  float acc = 0.f;
  for (int m = 0; m < EMB_; ++m) acc += W_eg[k * EMB_ + m] * Wg[m * OUT_ + d];
  WcT[d * DIN_ + k] = (_Float16)acc;
  if (k == 0) {
    float bb = 0.f;
    for (int m = 0; m < EMB_; ++m) bb += b_eg[m] * Wg[m * OUT_ + d];
    bc[d] = bb;
  }
}

// ---------------------------------------------------------------------------
// Kernel 1 body, PASS is compile-time so the epilogue is straight-line.
//   PASS 0: EA = traj@W_ge + b_ge  -> f16 row-major [token][e]
//   PASS 1: TH = traj@Wc   + bc    -> f16 TRANSPOSED [d][token]
// ---------------------------------------------------------------------------
template <int PASS>
__device__ __forceinline__ void embed_pass(
    const float* __restrict__ traj, const _Float16* __restrict__ WT,
    const float* __restrict__ bias_v, _Float16* __restrict__ dst,
    _Float16* Xs, _Float16* Ws, int b, int i0, int tid, int lane, int w) {
  const int n = lane & 15, hh = lane >> 4;
  v8f z = {};
  v8f acc[8];
#pragma unroll
  for (int nt = 0; nt < 8; ++nt) acc[nt] = z;

  for (int kc = 0; kc < 8; ++kc) {
    __syncthreads();
#pragma unroll
    for (int t = 0; t < 16; ++t) {
      int lin = tid * 16 + t;
      int row = lin >> 5, kk = lin & 31;
      Xs[lin] = (_Float16)traj[((size_t)b * L_ + i0 + row) * DIN_ + kc * 32 + kk];
      Ws[lin] = WT[row * DIN_ + kc * 32 + kk];
    }
    __syncthreads();
    const int arow = w * 16 + n;
    v16h a = ldsA(&Xs[arow * 32], hh ? 8 : 0);
    const int kbb = hh ? 16 : 0;
    // depth-2 pipelined B fragments
    v16h b0 = ldsB(&Ws[(0 * 16 + n) * 32 + kbb]);
#pragma unroll
    for (int nt = 0; nt < 8; ++nt) {
      v16h b1 = b0;
      if (nt < 7) b1 = ldsB(&Ws[((nt + 1) * 16 + n) * 32 + kbb]);
      acc[nt] = __builtin_amdgcn_wmma_f32_16x16x32_f16(
          false, a, false, b0, (short)0, acc[nt], false, false);
      b0 = b1;
    }
  }
#pragma unroll
  for (int nt = 0; nt < 8; ++nt) {
    const int col = nt * 16 + n;
    const float bias = bias_v[col];
#pragma unroll
    for (int r = 0; r < 8; ++r) {
      const int M = r + (hh ? 8 : 0);
      const int row = i0 + w * 16 + M;
      const float v = acc[nt][r] + bias;
      if (PASS == 0)
        dst[((size_t)b * L_ + row) * EMB_ + col] = (_Float16)v;
      else
        dst[((size_t)b * OUT_ + col) * L_ + row] = (_Float16)v;
    }
  }
}

__global__ __launch_bounds__(256) void embed_kernel(
    const float* __restrict__ traj,
    const _Float16* __restrict__ WgeT, const _Float16* __restrict__ WcT,
    const float* __restrict__ b_ge, const float* __restrict__ bc,
    _Float16* __restrict__ EAh, _Float16* __restrict__ THT) {
  extern __shared__ __align__(256) _Float16 sm[];
  _Float16* Xs = sm;              // 128 rows x 32 k
  _Float16* Ws = sm + 128 * 32;   // 128 n   x 32 k
  const int b = blockIdx.y, i0 = blockIdx.x * 128;
  const int tid = threadIdx.x, lane = tid & 31, w = tid >> 5;
  embed_pass<0>(traj, WgeT, b_ge, EAh, Xs, Ws, b, i0, tid, lane, w);
  embed_pass<1>(traj, WcT,  bc,   THT, Xs, Ws, b, i0, tid, lane, w);
}

// ---------------------------------------------------------------------------
// Kernel 2: fused relu(Q·K^T)*mask -> online softmax -> A @ theta -> LN*mask.
// Block = (batch, 128 query rows). 8 waves; each wave owns 16 rows.
// LDS: Q (32KB) | K-tile reused as P (32KB) | theta^T tile (32KB).
// ---------------------------------------------------------------------------
__global__ __launch_bounds__(256) void attn_kernel(
    const _Float16* __restrict__ EAh, const _Float16* __restrict__ THT,
    const int* __restrict__ traj_length,
    const float* __restrict__ gamma, const float* __restrict__ beta,
    float* __restrict__ out) {
  extern __shared__ __align__(256) _Float16 sm[];
  _Float16* Qs = sm;            // [128][128]
  _Float16* Ks = sm + 16384;    // [128][128]  K tile, then P tile
  _Float16* Ts = sm + 32768;    // [128][128]  theta^T: [d][j]
  const int b = blockIdx.y, i0 = blockIdx.x * 128;
  const int tid = threadIdx.x, lane = tid & 31, w = tid >> 5;
  const int n = lane & 15, hh = lane >> 4;
  const int len = traj_length[b];
  const int arow = w * 16 + n;
  const int kbA = hh ? 8 : 0;
  const int kbB = hh ? 16 : 0;

  // stage Q tile (contiguous 32KB)
  {
    const uint4* src = (const uint4*)(EAh + ((size_t)b * L_ + i0) * EMB_);
    uint4* dst = (uint4*)Qs;
#pragma unroll
    for (int t = 0; t < 8; ++t) dst[tid + t * 256] = src[tid + t * 256];
  }
  __syncthreads();

  // Q A-fragments are invariant across the whole j loop: hoist them.
  v16h qa[4];
#pragma unroll
  for (int kc = 0; kc < 4; ++kc) qa[kc] = ldsA(&Qs[arow * 128 + kc * 32], kbA);

  float m[8], l[8];
  v8f z = {};
  v8f O[8];
#pragma unroll
  for (int r = 0; r < 8; ++r) { m[r] = 0.f; l[r] = 0.f; }
#pragma unroll
  for (int dt = 0; dt < 8; ++dt) O[dt] = z;

  for (int jt = 0; jt < 8; ++jt) {
    const int j0 = jt * 128;
    __syncthreads();
    // stage K tile (EA rows j0..j0+127) and theta^T tile ([d][j0..j0+127])
    {
      const uint4* src = (const uint4*)(EAh + ((size_t)b * L_ + j0) * EMB_);
      uint4* dst = (uint4*)Ks;
#pragma unroll
      for (int t = 0; t < 8; ++t) dst[tid + t * 256] = src[tid + t * 256];
    }
    {
      const int row = tid >> 1, part = tid & 1;
      const uint4* src =
          (const uint4*)(THT + ((size_t)b * OUT_ + row) * L_ + j0 + part * 64);
      uint4* dst = (uint4*)(Ts + row * 128 + part * 64);
#pragma unroll
      for (int t = 0; t < 8; ++t) dst[t] = src[t];
    }
    if (jt < 7)  // prefetch next K tile into L2
      __builtin_prefetch(EAh + ((size_t)b * L_ + j0 + 128) * EMB_ + tid * 64, 0, 1);
    __syncthreads();

    // ---- S = Q @ K^T (wave: 16 rows x 128 cols) ----
    v8f S[8];
#pragma unroll
    for (int nt = 0; nt < 8; ++nt) S[nt] = z;
#pragma unroll
    for (int kc = 0; kc < 4; ++kc) {
      v16h b0 = ldsB(&Ks[(0 * 16 + n) * 128 + kc * 32 + kbB]);
#pragma unroll
      for (int nt = 0; nt < 8; ++nt) {
        v16h b1 = b0;
        if (nt < 7) b1 = ldsB(&Ks[((nt + 1) * 16 + n) * 128 + kc * 32 + kbB]);
        S[nt] = __builtin_amdgcn_wmma_f32_16x16x32_f16(
            false, qa[kc], false, b0, (short)0, S[nt], false, false);
        b0 = b1;
      }
    }

    // ---- relu, zero-mask (exp(0) still contributes!), online softmax ----
#pragma unroll
    for (int r = 0; r < 8; ++r) {
      float rm = 0.f;  // all values >= 0 after relu/mask
#pragma unroll
      for (int nt = 0; nt < 8; ++nt) {
        const int jg = j0 + nt * 16 + n;
        float s = S[nt][r];
        s = (jg < len) ? fmaxf(s, 0.f) : 0.f;
        S[nt][r] = s;
        rm = fmaxf(rm, s);
      }
      for (int xm = 1; xm < 16; xm <<= 1) rm = fmaxf(rm, __shfl_xor(rm, xm, 32));
      const float mn = fmaxf(m[r], rm);
      const float sc = __expf(m[r] - mn);
      float rs = 0.f;
#pragma unroll
      for (int nt = 0; nt < 8; ++nt) {
        const float p = __expf(S[nt][r] - mn);
        S[nt][r] = p;
        rs += p;
      }
      for (int xm = 1; xm < 16; xm <<= 1) rs += __shfl_xor(rs, xm, 32);
      l[r] = l[r] * sc + rs;
      m[r] = mn;
#pragma unroll
      for (int dt = 0; dt < 8; ++dt) O[dt][r] *= sc;
    }

    __syncthreads();  // all waves done reading K tile
    // write P (f16) into the K-tile LDS region (wave-private rows)
#pragma unroll
    for (int nt = 0; nt < 8; ++nt)
#pragma unroll
      for (int r = 0; r < 8; ++r) {
        const int M = r + (hh ? 8 : 0);
        Ks[(w * 16 + M) * 128 + nt * 16 + n] = (_Float16)S[nt][r];
      }
    __syncthreads();

    // ---- O += P @ theta^T ----
#pragma unroll
    for (int jc = 0; jc < 4; ++jc) {
      v16h a = ldsA(&Ks[arow * 128 + jc * 32], kbA);
      v16h b0 = ldsB(&Ts[(0 * 16 + n) * 128 + jc * 32 + kbB]);
#pragma unroll
      for (int dt = 0; dt < 8; ++dt) {
        v16h b1 = b0;
        if (dt < 7) b1 = ldsB(&Ts[((dt + 1) * 16 + n) * 128 + jc * 32 + kbB]);
        O[dt] = __builtin_amdgcn_wmma_f32_16x16x32_f16(
            false, a, false, b0, (short)0, O[dt], false, false);
        b0 = b1;
      }
    }
  }

  // ---- epilogue: 1/l, layernorm over 128 cols, row mask, fp32 store ----
#pragma unroll
  for (int r = 0; r < 8; ++r) {
    const float inv = 1.f / l[r];
    float mu = 0.f;
#pragma unroll
    for (int dt = 0; dt < 8; ++dt) { O[dt][r] *= inv; mu += O[dt][r]; }
    for (int xm = 1; xm < 16; xm <<= 1) mu += __shfl_xor(mu, xm, 32);
    mu *= (1.f / 128.f);
    float var = 0.f;
#pragma unroll
    for (int dt = 0; dt < 8; ++dt) {
      const float d = O[dt][r] - mu;
      var += d * d;
    }
    for (int xm = 1; xm < 16; xm <<= 1) var += __shfl_xor(var, xm, 32);
    var *= (1.f / 128.f);
    const float rstd = rsqrtf(var + 1e-5f);
    const int M = r + (hh ? 8 : 0);
    const int ig = i0 + w * 16 + M;
    const float rowmask = (ig < len) ? 1.f : 0.f;
#pragma unroll
    for (int dt = 0; dt < 8; ++dt) {
      const int col = dt * 16 + n;
      const float v = (O[dt][r] - mu) * rstd * gamma[col] + beta[col];
      out[((size_t)b * L_ + ig) * OUT_ + col] = v * rowmask;
    }
  }
}

// ---------------------------------------------------------------------------
extern "C" void kernel_launch(void* const* d_in, const int* in_sizes, int n_in,
                              void* d_out, int out_size, void* d_ws, size_t ws_size,
                              hipStream_t stream) {
  const float* traj        = (const float*)d_in[0];
  const int*   traj_length = (const int*)d_in[1];
  const float* W_ge        = (const float*)d_in[2];
  const float* b_ge        = (const float*)d_in[3];
  const float* W_eg        = (const float*)d_in[4];
  const float* b_eg        = (const float*)d_in[5];
  const float* Wg          = (const float*)d_in[6];
  const float* ln_gamma    = (const float*)d_in[7];
  const float* ln_beta     = (const float*)d_in[8];
  float* out = (float*)d_out;

  char* ws = (char*)d_ws;
  const size_t EA_BYTES = (size_t)B_ * L_ * EMB_ * 2;   // 8 MB
  _Float16* EAh  = (_Float16*)(ws);
  _Float16* THT  = (_Float16*)(ws + EA_BYTES);
  _Float16* WgeT = (_Float16*)(ws + 2 * EA_BYTES);
  _Float16* WcT  = (_Float16*)(ws + 2 * EA_BYTES + (size_t)EMB_ * DIN_ * 2);
  float*    bc   = (float*)   (ws + 2 * EA_BYTES + (size_t)2 * EMB_ * DIN_ * 2);

  prep_weights<<<dim3(DIN_), dim3(128), 0, stream>>>(W_ge, W_eg, b_eg, Wg,
                                                     WgeT, WcT, bc);
  embed_kernel<<<dim3(L_ / 128, B_), dim3(256),
                 2 * 128 * 32 * sizeof(_Float16), stream>>>(
      traj, WgeT, WcT, b_ge, bc, EAh, THT);
  attn_kernel<<<dim3(L_ / 128, B_), dim3(256),
                3 * 128 * 128 * sizeof(_Float16), stream>>>(
      EAh, THT, traj_length, ln_gamma, ln_beta, out);
}
